// OctreeConvTriton_33844342292892
// MI455X (gfx1250) — matrix operand order repaired and checked by reading, not compiled, not run on hardware
//
#include <hip/hip_runtime.h>

// Octree sparse conv for MI455X (gfx1250, wave32).
// out[m,:] = sum_k W_k^T(Cin x Cout) @ data[neigh[m,k],:]   (zero if neigh = -1)
//
// Compute-bound: 132.7 GFLOP vs ~170 MB HBM (~7 us at 23.3 TB/s; the 27x row
// gather is served by the 192 MB L2). bf16 hi/lo split (3 WMMAs / tile) gives
// ~fp32 accuracy at 3/8 the cost of the fp32 WMMA path.
//
// Weights are pre-split/pre-swizzled once into d_ws (1.77 MB); the main loop
// stages each 64 KB W_k block into LDS with a single tensor_load_to_lds DMA
// (TENSORcnt), double-buffered so the TDM copy of k+1 overlaps compute of k.

#define NPTS  150000
#define KOFF  27
#define CINC  128
#define COUTC 128
#define PLANE (COUTC * CINC)          // 16384 bf16 per plane
#define KBLK  (2 * PLANE)             // hi+lo per k: 32768 bf16 = 65536 B
#define WS_BYTES ((size_t)KOFF * KBLK * 2)

typedef __bf16 bf16;
typedef bf16  v16bf __attribute__((ext_vector_type(16)));
typedef bf16  v8bf  __attribute__((ext_vector_type(8)));
typedef float v8f   __attribute__((ext_vector_type(8)));
typedef float v4f   __attribute__((ext_vector_type(4)));
typedef unsigned int v4u __attribute__((ext_vector_type(4)));
typedef int   v8i   __attribute__((ext_vector_type(8)));
typedef int   v4i   __attribute__((ext_vector_type(4)));

struct bf2 { bf16 h; bf16 l; };

__device__ __forceinline__ bf2 split_bf16(float x) {
    bf2 r;
    bf16 hb = (bf16)x;           // RNE f32 -> bf16
    r.h = hb;
    r.l = (bf16)(x - (float)hb); // residual in bf16
    return r;
}

// Swizzled B-operand offset: rows of 128 bf16 (256B), 16B chunks XORed with
// (cout & 15) -> conflict-free ds_load_b128 within each half-wave.
__device__ __forceinline__ int bswz(int cout, int cin) {
    return cout * CINC + (((cin >> 3) ^ (cout & 15)) << 3) + (cin & 7);
}

// ---------------------------------------------------------------------------
// One-time weight pack: f32 [Cout,K,Cin] -> per-k contiguous 64KB blocks of
// bf16 {hi plane, lo plane} already in the swizzled B layout.
// ---------------------------------------------------------------------------
__global__ __launch_bounds__(256) void pack_weights(
    const float* __restrict__ weights, bf16* __restrict__ wpack)
{
    const int k     = blockIdx.x;
    const int tid   = threadIdx.x;
    const int wrow  = tid >> 1;          // cout
    const int wcin0 = (tid & 1) * 64;    // cin base
    const float* wsrc = weights + wrow * (KOFF * CINC) + k * CINC + wcin0;
    bf16* dst = wpack + (size_t)k * KBLK;
#pragma unroll
    for (int i = 0; i < 16; ++i) {
        const int cin = wcin0 + i * 4;
        v4f wv = *(const v4f*)(wsrc + i * 4);
        const int off = bswz(wrow, cin);
#pragma unroll
        for (int j = 0; j < 4; ++j) {
            bf2 s = split_bf16(wv[j]);
            dst[off + j]         = s.h;
            dst[PLANE + off + j] = s.l;
        }
    }
}

// ---------------------------------------------------------------------------
// TDM descriptor issue: copy one contiguous 64KB W_k block -> LDS.
// 1D tensor: 32768 x 2B elements, tile = whole tensor. (D# per ISA ch.8)
// This toolchain's builtin is the 6-arg form: (g0, g1, g2, g3, g4, cpol).
// ---------------------------------------------------------------------------
__device__ __forceinline__ void tdm_load_wk(const bf16* gsrc, unsigned lds_off) {
    const unsigned long long ga = (unsigned long long)(size_t)gsrc;
    v4u g0;
    g0[0] = 1u;                                              // count=1, user desc
    g0[1] = lds_off;                                         // lds_addr
    g0[2] = (unsigned)(ga & 0xFFFFFFFFu);                    // global_addr[31:0]
    g0[3] = (unsigned)((ga >> 32) & 0x01FFFFFFu) | (2u << 30); // [56:32] | type=2
    v8i g1;
    g1[0] = 0x00010000;        // wg_mask=0, data_size=1 (2B), no barrier/iter/pad
    g1[1] = (int)0x80000000u;  // tensor_dim0 = 0x8000 (bits 79:48 -> [63:48])
    g1[2] = 0x00010000;        // tensor_dim0 msbs = 0; tensor_dim1 = 1
    g1[3] = (int)0x80000000u;  // tile_dim0 = 0x8000 (bits 127:112)
    g1[4] = 0x00000001;        // tile_dim1 = 1, tile_dim2 = 0
    g1[5] = 0x00008000;        // tensor_dim0_stride = 0x8000
    g1[6] = (int)0x80000000u;  // tensor_dim1_stride = 0x8000 (low 16 in [31:16])
    g1[7] = 0;
    v4i gz4 = {0, 0, 0, 0};
    v8i gz8 = {0, 0, 0, 0, 0, 0, 0, 0};
    __builtin_amdgcn_tensor_load_to_lds(g0, g1, gz4, gz4, gz8, 0);
}

// ---------------------------------------------------------------------------
// Main kernel. USE_TDM=true : read pre-packed weights from d_ws via TDM,
//              USE_TDM=false: legacy in-kernel f32->bf16 staging fallback.
// ---------------------------------------------------------------------------
template <bool USE_TDM>
__global__ __launch_bounds__(256) void octree_conv_wmma(
    const float* __restrict__ data,       // [N, 128] f32
    const float* __restrict__ weights,    // [128, 27, 128] f32 (fallback path)
    const bf16*  __restrict__ wpack,      // [27] x 64KB packed blocks (TDM path)
    const int*   __restrict__ nbr32,      // [N, 27] i64 viewed as i32 pairs
    float* __restrict__ out)              // [N, 128] f32
{
    __shared__ bf16 smem[(USE_TDM ? 2 : 1) * KBLK];  // 128 KB / 64 KB of 320 KB

    const int tid   = threadIdx.x;
    const int lane  = tid & 31;
    const int wave  = tid >> 5;       // 0..7: which 16-row M sub-tile
    const int n     = lane & 15;      // column within a 16-wide tile
    const int h     = lane >> 4;      // wave half (WMMA operand layout selector)
    const int mbase = blockIdx.x * 128;
    const int arow  = mbase + wave * 16 + n;   // row this lane gathers for A

    const v8f zero8 = {0.f, 0.f, 0.f, 0.f, 0.f, 0.f, 0.f, 0.f};
    v8f acc[8];
#pragma unroll
    for (int t = 0; t < 8; ++t) acc[t] = zero8;

    unsigned lds_base = 0;
    if constexpr (USE_TDM) {
        lds_base = (unsigned)(size_t)&smem[0];   // LDS aperture: low 32 = offset
        if (wave == 0) {
            tdm_load_wk(wpack, lds_base);        // prologue: W_0 -> buf0
            __builtin_amdgcn_s_wait_tensorcnt(0);
        }
        __syncthreads();
    }

#pragma unroll 1
    for (int k = 0; k < KOFF; ++k) {
        int bufo = 0;
        if constexpr (USE_TDM) {
            bufo = (k & 1) * KBLK;
            // async-prefetch next k's weights into the other buffer
            if (wave == 0 && k + 1 < KOFF)
                tdm_load_wk(wpack + (size_t)(k + 1) * KBLK,
                            lds_base + (unsigned)(((k + 1) & 1) * KBLK) * 2u);
        } else {
            // ---- legacy: stage + split W_k into LDS ----
            const int wrow  = tid >> 1;
            const int wcin0 = (tid & 1) * 64;
            const float* wsrc = weights + wrow * (KOFF * CINC) + k * CINC + wcin0;
            if (k + 1 < KOFF) __builtin_prefetch(wsrc + CINC, 0, 1);
            __syncthreads();  // previous iteration's LDS reads done
#pragma unroll
            for (int i = 0; i < 16; ++i) {
                const int cin = wcin0 + i * 4;
                v4f wv = *(const v4f*)(wsrc + i * 4);
                const int off = bswz(wrow, cin);
#pragma unroll
                for (int j = 0; j < 4; ++j) {
                    bf2 s = split_bf16(wv[j]);
                    smem[off + j]         = s.h;
                    smem[PLANE + off + j] = s.l;
                }
            }
            __syncthreads();
        }

        // ---- gather this lane's A row for offset k (EXEC stays all-1s) ----
        // neighbour is int64 but values fit in the low dword (incl. -1 sign).
        int idx = -1;
        if (arow < NPTS) idx = nbr32[(arow * KOFF + k) * 2];
        const bool valid = (idx >= 0);
        const float* src = data + (size_t)(valid ? idx : 0) * CINC;

        const v4f z4 = {0.f, 0.f, 0.f, 0.f};
#pragma unroll
        for (int c0 = 0; c0 < CINC; c0 += 32) {
            // A 16x32 bf16 layout: lane(h,r): elems 0..7  -> cin c0+h*8+e,
            //                                 elems 8..15 -> cin c0+16+h*8+(e-8)
            v4f f0 = *(const v4f*)(src + c0 + h * 8);
            v4f f1 = *(const v4f*)(src + c0 + h * 8 + 4);
            v4f f2 = *(const v4f*)(src + c0 + 16 + h * 8);
            v4f f3 = *(const v4f*)(src + c0 + 16 + h * 8 + 4);
            if (!valid) { f0 = z4; f1 = z4; f2 = z4; f3 = z4; }  // cndmask

            v16bf a_hi, a_lo;
#pragma unroll
            for (int j = 0; j < 4; ++j) {
                bf2 s0 = split_bf16(f0[j]);
                bf2 s1 = split_bf16(f1[j]);
                bf2 s2 = split_bf16(f2[j]);
                bf2 s3 = split_bf16(f3[j]);
                a_hi[j]      = s0.h;  a_lo[j]      = s0.l;
                a_hi[4 + j]  = s1.h;  a_lo[4 + j]  = s1.l;
                a_hi[8 + j]  = s2.h;  a_lo[8 + j]  = s2.l;
                a_hi[12 + j] = s3.h;  a_lo[12 + j] = s3.l;
            }

#pragma unroll
            for (int t = 0; t < 8; ++t) {
                // B 32x16 bf16: lane(h,n): elems 0..7 -> cin c0+h*8+e, elems 8..15 -> cin c0+16+h*8+(e-8), cout t*16+n
                const int row = t * 16 + n;
                const int cb  = (c0 >> 3) + h;
                const int o0  = bufo + row * CINC + (((cb)     ^ n) << 3);
                const int o1  = bufo + row * CINC + (((cb + 2) ^ n) << 3);
                v8bf bh0 = *(const v8bf*)&smem[o0];
                v8bf bh1 = *(const v8bf*)&smem[o1];
                v8bf bl0 = *(const v8bf*)&smem[PLANE + o0];
                v8bf bl1 = *(const v8bf*)&smem[PLANE + o1];
                v16bf bh = __builtin_shufflevector(bh0, bh1, 0, 1, 2, 3, 4, 5, 6, 7,
                                                   8, 9, 10, 11, 12, 13, 14, 15);
                v16bf bl = __builtin_shufflevector(bl0, bl1, 0, 1, 2, 3, 4, 5, 6, 7,
                                                   8, 9, 10, 11, 12, 13, 14, 15);
                // bf16x3: hi*hi + hi*lo + lo*hi, f32 accumulate
                acc[t] = __builtin_amdgcn_wmma_f32_16x16x32_bf16(
                    false, a_hi, false, bh, (short)0, acc[t], false, false);
                acc[t] = __builtin_amdgcn_wmma_f32_16x16x32_bf16(
                    false, a_hi, false, bl, (short)0, acc[t], false, false);
                acc[t] = __builtin_amdgcn_wmma_f32_16x16x32_bf16(
                    false, a_lo, false, bh, (short)0, acc[t], false, false);
            }
        }

        if constexpr (USE_TDM) {
            __syncthreads();                                // all done reading buf k
            if (wave == 0) __builtin_amdgcn_s_wait_tensorcnt(0);  // k+1 DMA landed
            __syncthreads();                                // publish buf k+1
        }
    }

    // ---- write back: C/D layout lane(h,n), VGPR v -> row h*8+v, col t*16+n ----
#pragma unroll
    for (int t = 0; t < 8; ++t) {
#pragma unroll
        for (int v = 0; v < 8; ++v) {
            const int mrow = mbase + wave * 16 + h * 8 + v;
            if (mrow < NPTS) out[mrow * COUTC + t * 16 + n] = acc[t][v];
        }
    }
}

extern "C" void kernel_launch(void* const* d_in, const int* in_sizes, int n_in,
                              void* d_out, int out_size, void* d_ws, size_t ws_size,
                              hipStream_t stream) {
    (void)in_sizes; (void)n_in; (void)out_size;
    const float* data      = (const float*)d_in[0];
    const float* weights   = (const float*)d_in[1];
    const int*   nbr32     = (const int*)d_in[2];   // int64 viewed as dword pairs
    float*       out       = (float*)d_out;

    dim3 grid((NPTS + 127) / 128);  // 1172 workgroups, 128-row M tiles
    dim3 block(256);                // 8 wave32s

    if (ws_size >= WS_BYTES && d_ws != nullptr) {
        bf16* wpack = (bf16*)d_ws;
        pack_weights<<<dim3(KOFF), block, 0, stream>>>(weights, wpack);
        octree_conv_wmma<true><<<grid, block, 0, stream>>>(
            data, weights, wpack, nbr32, out);
    } else {
        octree_conv_wmma<false><<<grid, block, 0, stream>>>(
            data, weights, nullptr, nbr32, out);
    }
}
